// H_ATT_16372415332364
// MI455X (gfx1250) — matrix operand order repaired and checked by reading, not compiled, not run on hardware
//
#include <hip/hip_runtime.h>
#include <stdint.h>

typedef __attribute__((ext_vector_type(8)))  float  v8f;
typedef __attribute__((ext_vector_type(8)))  __bf16 v8bf;
typedef __attribute__((ext_vector_type(16))) __bf16 v16bf;

#define HDIM 1024
#define INDIM 2048
#define MROWS 2048   /* B*R */
#define RR 32
#define BB 64

#define BWMMA(A, Bm, C) \
    __builtin_amdgcn_wmma_f32_16x16x32_bf16(false, (A), false, (Bm), (short)0, (C), false, false)

static __device__ __forceinline__ v16bf ldfrag(const __bf16* p, int second) {
    v8bf a = *(const v8bf*)(p);
    v8bf b = *(const v8bf*)(p + second);
    return __builtin_shufflevector(a, b, 0, 1, 2, 3, 4, 5, 6, 7,
                                   8, 9, 10, 11, 12, 13, 14, 15);
}

static __device__ __forceinline__ void split_bf16(float f, __bf16* hi, __bf16* lo) {
    __bf16 h = (__bf16)f;
    *hi = h;
    *lo = (__bf16)(f - (float)h);
}

// 16B global -> LDS async copy (ASYNCcnt-tracked, no VGPR staging)
static __device__ __forceinline__ void async_copy16(uint32_t lds_off, uint64_t gaddr) {
    asm volatile("global_load_async_to_lds_b128 %0, %1, off"
                 :: "v"(lds_off), "v"(gaddr) : "memory");
}
static __device__ __forceinline__ void wait_asynccnt_le2() {
    asm volatile("s_wait_asynccnt 0x2" ::: "memory");
}

// ---------------------------------------------------------------------------
// Kernel 0a: elementwise split X (hist/ques) -> bf16 hi/lo, same [M][K] layout
// ---------------------------------------------------------------------------
__global__ __launch_bounds__(256) void convert_x_kernel(
    const float* __restrict__ hist, const float* __restrict__ ques,
    __bf16* __restrict__ xh_hi, __bf16* __restrict__ xh_lo,
    __bf16* __restrict__ xq_hi, __bf16* __restrict__ xq_lo)
{
    const float* X = blockIdx.y ? ques : hist;
    __bf16* hi = blockIdx.y ? xq_hi : xh_hi;
    __bf16* lo = blockIdx.y ? xq_lo : xh_lo;
    const size_t i = ((size_t)blockIdx.x * 256 + threadIdx.x) * 4;
    float4 v = *(const float4*)(X + i);
    __bf16 h, l;
    split_bf16(v.x, &h, &l); hi[i + 0] = h; lo[i + 0] = l;
    split_bf16(v.y, &h, &l); hi[i + 1] = h; lo[i + 1] = l;
    split_bf16(v.z, &h, &l); hi[i + 2] = h; lo[i + 2] = l;
    split_bf16(v.w, &h, &l); hi[i + 3] = h; lo[i + 3] = l;
}

// ---------------------------------------------------------------------------
// Kernel 0b: transpose-split W [K][N] -> Wt hi/lo bf16 [N][K] (LDS 32x32 tile)
// ---------------------------------------------------------------------------
__global__ __launch_bounds__(256) void convert_w_kernel(
    const float* __restrict__ W_hy, const float* __restrict__ W_hg,
    const float* __restrict__ W_qy, const float* __restrict__ W_qg,
    __bf16* __restrict__ wty_h_hi, __bf16* __restrict__ wty_h_lo,
    __bf16* __restrict__ wtg_h_hi, __bf16* __restrict__ wtg_h_lo,
    __bf16* __restrict__ wty_q_hi, __bf16* __restrict__ wty_q_lo,
    __bf16* __restrict__ wtg_q_hi, __bf16* __restrict__ wtg_q_lo)
{
    __shared__ float tile[32][33];
    const float* W;
    __bf16 *Whi, *Wlo;
    switch (blockIdx.z) {
        case 0:  W = W_hy; Whi = wty_h_hi; Wlo = wty_h_lo; break;
        case 1:  W = W_hg; Whi = wtg_h_hi; Wlo = wtg_h_lo; break;
        case 2:  W = W_qy; Whi = wty_q_hi; Wlo = wty_q_lo; break;
        default: W = W_qg; Whi = wtg_q_hi; Wlo = wtg_q_lo; break;
    }
    const int k0 = blockIdx.x * 32;
    const int n0 = blockIdx.y * 32;
    const int j  = threadIdx.x & 31;
    const int i0 = threadIdx.x >> 5;
#pragma unroll
    for (int r = 0; r < 4; ++r) {
        int i = i0 + r * 8;
        tile[i][j] = W[(size_t)(k0 + i) * HDIM + (n0 + j)];
    }
    __syncthreads();
#pragma unroll
    for (int r = 0; r < 4; ++r) {
        int i = i0 + r * 8;                      // n offset within tile
        float f = tile[j][i];                    // = W[k0+j][n0+i]
        __bf16 h, l;
        split_bf16(f, &h, &l);
        size_t o = (size_t)(n0 + i) * INDIM + (k0 + j);
        Whi[o] = h;
        Wlo[o] = l;
    }
}

// ---------------------------------------------------------------------------
// Kernel 1: E = tanh(X@Wy + by) * leaky_relu(X@Wg + bg), bf16x3 split GEMM.
// Block = 8 waves covering 64(M) x 32(N); per K=32 chunk the block's 8KB of
// weight fragments are staged into LDS with global_load_async_to_lds_b128
// (double-buffered, s_wait_asynccnt + barrier), then each wave runs 6 bf16
// WMMAs (hi*hi, hi*lo, lo*hi per branch).  blockIdx.z: 0=hist->he, 1=ques->qe
// ---------------------------------------------------------------------------
__global__ __launch_bounds__(256) void gated_embed_bf16x3(
    const __bf16* __restrict__ xh_hi, const __bf16* __restrict__ xh_lo,
    const __bf16* __restrict__ xq_hi, const __bf16* __restrict__ xq_lo,
    const __bf16* __restrict__ wty_h_hi, const __bf16* __restrict__ wty_h_lo,
    const __bf16* __restrict__ wtg_h_hi, const __bf16* __restrict__ wtg_h_lo,
    const __bf16* __restrict__ wty_q_hi, const __bf16* __restrict__ wty_q_lo,
    const __bf16* __restrict__ wtg_q_hi, const __bf16* __restrict__ wtg_q_lo,
    const float* __restrict__ b_hy, const float* __restrict__ b_hg,
    const float* __restrict__ b_qy, const float* __restrict__ b_qg,
    float* __restrict__ he, float* __restrict__ qe)
{
    // double-buffered weight stage: [buf][arr(4)][row(32)][64B]
    __shared__ __align__(32) char bbuf[2][4 * 32 * 64];

    const int lane = threadIdx.x & 31;
    const int wave = threadIdx.x >> 5;
    const bool isq = (blockIdx.z != 0);
    const __bf16* Ah  = isq ? xq_hi : xh_hi;
    const __bf16* Al  = isq ? xq_lo : xh_lo;
    const __bf16* Byh = isq ? wty_q_hi : wty_h_hi;
    const __bf16* Byl = isq ? wty_q_lo : wty_h_lo;
    const __bf16* Bgh = isq ? wtg_q_hi : wtg_h_hi;
    const __bf16* Bgl = isq ? wtg_q_lo : wtg_h_lo;
    const float* by = isq ? b_qy : b_hy;
    const float* bg = isq ? b_qg : b_hg;
    float* E = isq ? qe : he;

    const int tileNblk = blockIdx.x * 32;
    const int tileM = blockIdx.y * 64 + (wave & 3) * 16;
    const int tileN = tileNblk + (wave >> 2) * 16;

    // ---- async-copy assignments: 512 x 16B per chunk, 2 per thread ----
    // id = arr*128 + row*4 + hp  (arr: y_hi,y_lo,g_hi,g_lo; row: n-tileNblk)
    const uint32_t ldsbase = (uint32_t)(uintptr_t)&bbuf[0][0];
    const __bf16* gsrc[2];
    uint32_t loff[2];
#pragma unroll
    for (int p = 0; p < 2; ++p) {
        int id  = wave * 64 + p * 32 + lane;
        int arr = id >> 7;
        int rem = id & 127;
        int row = rem >> 2;
        int hp  = rem & 3;
        const __bf16* wb = (arr == 0) ? Byh : (arr == 1) ? Byl
                          : (arr == 2) ? Bgh : Bgl;
        gsrc[p] = wb + (size_t)(tileNblk + row) * INDIM + hp * 8;
        loff[p] = (uint32_t)(arr * 2048 + row * 64 + hp * 16);
    }

    // A 16x32 bf16 fragment: lanes 0-15 -> M=lane, K {0..15};
    //                        lanes 16-31 -> M=lane-16, K {16..31}
    const int mA   = tileM + (lane & 15);
    const int aoff = (lane >> 4) * 16;
    const __bf16* arow_hi = Ah + (size_t)mA * INDIM + aoff;
    const __bf16* arow_lo = Al + (size_t)mA * INDIM + aoff;

    // B fragment from LDS: per lane 32 contiguous bytes
    const int rowF = (wave >> 2) * 16 + (lane & 15);
    const uint32_t fo = (uint32_t)(rowF * 64 + (lane >> 4) * 32);

    // prologue: stage chunk 0 into buffer 0
#pragma unroll
    for (int p = 0; p < 2; ++p)
        async_copy16(ldsbase + loff[p], (uint64_t)(uintptr_t)gsrc[p]);

    v8f cy = {};
    v8f cg = {};
    for (int kc = 0; kc < 64; ++kc) {
        const int k = kc * 32;
        __syncthreads();   // everyone done reading the buffer we overwrite next
        const int knext = ((kc + 1) & 63) * 32;
        const uint32_t nbuf = ldsbase + (uint32_t)(((kc + 1) & 1) * 8192);
#pragma unroll
        for (int p = 0; p < 2; ++p)
            async_copy16(nbuf + loff[p], (uint64_t)(uintptr_t)(gsrc[p] + knext));
        wait_asynccnt_le2();   // current chunk's copies landed (in-order)
        __syncthreads();       // ... for every wave in the block

        __builtin_prefetch(arow_hi + k + 256, 0, 3);
        __builtin_prefetch(arow_lo + k + 256, 0, 3);
        v16bf ahi = ldfrag(arow_hi + k, 8);
        v16bf alo = ldfrag(arow_lo + k, 8);

        const char* bb = (const char*)&bbuf[kc & 1][0];
        v16bf byh = *(const v16bf*)(bb + 0    + fo);
        v16bf byl = *(const v16bf*)(bb + 2048 + fo);
        v16bf bgh = *(const v16bf*)(bb + 4096 + fo);
        v16bf bgl = *(const v16bf*)(bb + 6144 + fo);

        cy = BWMMA(ahi, byh, cy);
        cy = BWMMA(ahi, byl, cy);
        cy = BWMMA(alo, byh, cy);
        cg = BWMMA(ahi, bgh, cg);
        cg = BWMMA(ahi, bgl, cg);
        cg = BWMMA(alo, bgh, cg);
    }

    // D layout: VGPR r -> lanes 0-15: M=r, lanes 16-31: M=r+8; N=lane&15
    const int mD = tileM + ((lane >> 4) << 3);
    const int nD = tileN + (lane & 15);
    const float biasY = by[nD];
    const float biasG = bg[nD];
#pragma unroll
    for (int r = 0; r < 8; ++r) {
        float y = cy[r] + biasY;
        float g = cg[r] + biasG;
        float lr = (g > 0.0f) ? g : 0.01f * g;
        E[(size_t)(mD + r) * HDIM + nD] = tanhf(y) * lr;
    }
}

// ---------------------------------------------------------------------------
// Kernel 2: normalized scores + causal-masked softmax.
// One wave per (b,q). softmax->tril->renorm == masked softmax over h<=q.
// ---------------------------------------------------------------------------
__global__ __launch_bounds__(256) void att_softmax_kernel(
    const float* __restrict__ he, const float* __restrict__ qe,
    const float* __restrict__ W_att, const float* __restrict__ b_att_p,
    float* __restrict__ att)
{
    const int lane = threadIdx.x & 31;
    const int wave = threadIdx.x >> 5;
    const int pair = blockIdx.x * 8 + wave;   // b*32 + q
    const int b = pair >> 5;
    const int q = pair & 31;
    const float b_att = b_att_p[0];

    float qreg[32], wreg[32];
    const float* qrow = qe + (size_t)pair * HDIM;
#pragma unroll
    for (int i = 0; i < 32; ++i) {
        qreg[i] = qrow[lane + 32 * i];
        wreg[i] = W_att[lane + 32 * i];
    }

    float sreg = 0.0f;
    for (int h = 0; h < 32; ++h) {
        const float* hrow = he + (size_t)(b * 32 + h) * HDIM;
        float s1 = 0.0f, s2 = 0.0f;
#pragma unroll
        for (int i = 0; i < 32; ++i) {
            float p = qreg[i] * hrow[lane + 32 * i];
            s1 = fmaf(p, wreg[i], s1);
            s2 = fmaf(p, p, s2);
        }
#pragma unroll
        for (int off = 16; off > 0; off >>= 1) {
            s1 += __shfl_xor(s1, off, 32);
            s2 += __shfl_xor(s2, off, 32);
        }
        float s = s1 / fmaxf(sqrtf(s2), 1e-12f) + b_att;
        if (lane == h) sreg = s;
    }

    float m = sreg;
#pragma unroll
    for (int off = 16; off > 0; off >>= 1)
        m = fmaxf(m, __shfl_xor(m, off, 32));
    float e = __expf(sreg - m);
    float val = (lane <= q) ? e : 0.0f;
    float sum = val;
#pragma unroll
    for (int off = 16; off > 0; off >>= 1)
        sum += __shfl_xor(sum, off, 32);
    att[(size_t)pair * 32 + lane] = val / sum;
}

// ---------------------------------------------------------------------------
// Kernel 3: feat[b,q,:] = sum_h att[b,q,h] * hist[b,h,:]
// ---------------------------------------------------------------------------
__global__ __launch_bounds__(256) void feat_kernel(
    const float* __restrict__ att, const float* __restrict__ hist,
    float* __restrict__ out)
{
    __shared__ float att_s[32 * 32];
    const int b = blockIdx.y;
    const int tid = threadIdx.x;
#pragma unroll
    for (int i = 0; i < 4; ++i)
        att_s[tid + 256 * i] = att[(size_t)b * 1024 + tid + 256 * i];
    __syncthreads();

    const int d = blockIdx.x * 256 + tid;
    const float* hb = hist + (size_t)b * 32 * INDIM + d;
    float hreg[32];
#pragma unroll
    for (int h = 0; h < 32; ++h) hreg[h] = hb[(size_t)h * INDIM];

    for (int q = 0; q < 32; ++q) {
        float acc = 0.0f;
#pragma unroll
        for (int h = 0; h < 32; ++h)
            acc = fmaf(att_s[q * 32 + h], hreg[h], acc);
        out[(size_t)(b * 32 + q) * INDIM + d] = acc;
    }
}

// ---------------------------------------------------------------------------
extern "C" void kernel_launch(void* const* d_in, const int* in_sizes, int n_in,
                              void* d_out, int out_size, void* d_ws, size_t ws_size,
                              hipStream_t stream) {
    const float* hist  = (const float*)d_in[0];
    const float* ques  = (const float*)d_in[1];
    const float* W_hy  = (const float*)d_in[2];
    const float* b_hy  = (const float*)d_in[3];
    const float* W_hg  = (const float*)d_in[4];
    const float* b_hg  = (const float*)d_in[5];
    const float* W_qy  = (const float*)d_in[6];
    const float* b_qy  = (const float*)d_in[7];
    const float* W_qg  = (const float*)d_in[8];
    const float* b_qg  = (const float*)d_in[9];
    const float* W_att = (const float*)d_in[10];
    const float* b_att = (const float*)d_in[11];
    float* out = (float*)d_out;

    // ---- workspace layout (bytes) ----
    char* wsb = (char*)d_ws;
    const size_t MB = 1024u * 1024u;
    float* he  = (float*)(wsb);                 // 8 MB
    float* qe  = (float*)(wsb + 8 * MB);        // 8 MB
    float* att = (float*)(wsb + 16 * MB);       // 256 KB
    __bf16* xh_hi = (__bf16*)(wsb + 17 * MB);   // 8 MB each
    __bf16* xh_lo = (__bf16*)(wsb + 25 * MB);
    __bf16* xq_hi = (__bf16*)(wsb + 33 * MB);
    __bf16* xq_lo = (__bf16*)(wsb + 41 * MB);
    __bf16* wty_h_hi = (__bf16*)(wsb + 49 * MB);  // 4 MB each
    __bf16* wty_h_lo = (__bf16*)(wsb + 53 * MB);
    __bf16* wtg_h_hi = (__bf16*)(wsb + 57 * MB);
    __bf16* wtg_h_lo = (__bf16*)(wsb + 61 * MB);
    __bf16* wty_q_hi = (__bf16*)(wsb + 65 * MB);
    __bf16* wty_q_lo = (__bf16*)(wsb + 69 * MB);
    __bf16* wtg_q_hi = (__bf16*)(wsb + 73 * MB);
    __bf16* wtg_q_lo = (__bf16*)(wsb + 77 * MB);

    convert_x_kernel<<<dim3((MROWS * INDIM) / (256 * 4), 2), 256, 0, stream>>>(
        hist, ques, xh_hi, xh_lo, xq_hi, xq_lo);

    convert_w_kernel<<<dim3(INDIM / 32, HDIM / 32, 4), 256, 0, stream>>>(
        W_hy, W_hg, W_qy, W_qg,
        wty_h_hi, wty_h_lo, wtg_h_hi, wtg_h_lo,
        wty_q_hi, wty_q_lo, wtg_q_hi, wtg_q_lo);

    gated_embed_bf16x3<<<dim3(HDIM / 32, MROWS / 64, 2), 256, 0, stream>>>(
        xh_hi, xh_lo, xq_hi, xq_lo,
        wty_h_hi, wty_h_lo, wtg_h_hi, wtg_h_lo,
        wty_q_hi, wty_q_lo, wtg_q_hi, wtg_q_lo,
        b_hy, b_hg, b_qy, b_qg, he, qe);

    att_softmax_kernel<<<dim3((BB * RR) / 8), 256, 0, stream>>>(
        he, qe, W_att, b_att, att);

    feat_kernel<<<dim3(INDIM / 256, BB), 256, 0, stream>>>(att, hist, out);
}
